// DPASSMBlock_65051574665460
// MI455X (gfx1250) — matrix operand order repaired and checked by reading, not compiled, not run on hardware
//
#include <hip/hip_runtime.h>
#include <math.h>

typedef __bf16 bf16_t;
typedef __attribute__((ext_vector_type(16))) __bf16 v16bf;
typedef __attribute__((ext_vector_type(8)))  float  v8f;

#define DEV static __device__ __forceinline__

// Model dims (fixed by the reference)
constexpr int kB  = 2,  kT = 2048, kD = 1024, kH = 16, kDH = 64;
constexpr int kW  = 256, kNB = 8,  kN = 256;
constexpr int kBT = kB * kT;          // 4096 token rows
constexpr int kDFF = 4 * kD;          // 4096
constexpr int kW2 = 2 * kW;           // 512 keys per window
constexpr int kZ  = kB * kH * kNB;    // 256 (b,h,block) tiles

constexpr int LDSSTRIDE = 40;         // padded bf16 row stride (80B, 16B-aligned, conflict-free)

union FragU { uint4 q[2]; v16bf v; };

DEV v8f wmma_bf16(v16bf a, v16bf b, v8f c) {
  return __builtin_amdgcn_wmma_f32_16x16x32_bf16(false, a, false, b, (short)0, c, false, false);
}

// A fragment (16x32 bf16): lane -> row M=lane&15, K = kOff + (laneHi*16) + [0..15]
DEV v16bf frag_a_ptr(const bf16_t* p0, long long rowStride, int row0, int kOff, int lane) {
  const bf16_t* p = p0 + (long long)(row0 + (lane & 15)) * rowStride + kOff + ((lane >> 4) << 4);
  FragU f;
  f.q[0] = *reinterpret_cast<const uint4*>(p);
  f.q[1] = *reinterpret_cast<const uint4*>(p + 8);
  return f.v;
}

// B fragment (32x16 bf16, stored as N x K rows): lane -> col N=lane&15, K = kOff + (laneHi*16) + [0..15]
DEV v16bf frag_b_ptr(const bf16_t* p0, long long rowStride, int row0, int kOff, int lane) {
  const bf16_t* p = p0 + (long long)(row0 + (lane & 15)) * rowStride + kOff + ((lane >> 4) << 4);
  FragU f;
  f.q[0] = *reinterpret_cast<const uint4*>(p);
  f.q[1] = *reinterpret_cast<const uint4*>(p + 8);
  return f.v;
}

// B fragment with per-row zero predicate (rows < minRow read as zero)
DEV v16bf frag_b_rowpred(const bf16_t* p0, long long rowStride, int row0, int kOff, int lane, int minRow) {
  int r = row0 + (lane & 15);
  FragU f;
  if (r >= minRow) {
    const bf16_t* p = p0 + (long long)r * rowStride + kOff + ((lane >> 4) << 4);
    f.q[0] = *reinterpret_cast<const uint4*>(p);
    f.q[1] = *reinterpret_cast<const uint4*>(p + 8);
  } else {
    f.q[0] = make_uint4(0u, 0u, 0u, 0u);
    f.q[1] = make_uint4(0u, 0u, 0u, 0u);
  }
  return f.v;
}

// ---------------------------------------------------------------------------
// CDNA5 async global->LDS staging (ASYNCcnt-tracked, no VGPR data round-trip).
// INST_OFFSET is added to BOTH the global and LDS addresses, so one address
// pair stages 64 contiguous bytes with 4 b128 transfers.
// ---------------------------------------------------------------------------
DEV void async_stage_64B(const bf16_t* gsrc, unsigned ldsDst) {
  unsigned long long ga = (unsigned long long)(size_t)gsrc;
  asm volatile(
      "global_load_async_to_lds_b128 %0, %1, off\n\t"
      "global_load_async_to_lds_b128 %0, %1, off offset:16\n\t"
      "global_load_async_to_lds_b128 %0, %1, off offset:32\n\t"
      "global_load_async_to_lds_b128 %0, %1, off offset:48"
      :: "v"(ldsDst), "v"(ga)
      : "memory");
}

DEV void async_wait0() {
  asm volatile("s_wait_asynccnt 0x0" ::: "memory");
}

// ---------------------------------------------------------------------------
// Weight convert + transpose: fp32 W[K,N] -> bf16 Wt[N,K]
// ---------------------------------------------------------------------------
__global__ void convert_wt_kernel(const float* __restrict__ W, bf16_t* __restrict__ Wt, int K, int N) {
  size_t idx = (size_t)blockIdx.x * 256 + threadIdx.x;
  if (idx >= (size_t)K * N) return;
  int n = (int)(idx % N);
  int k = (int)(idx / N);
  Wt[(size_t)n * K + k] = (bf16_t)W[idx];
}

// V [B,T,H,DH] bf16 -> Vt [B,H,DH,T] bf16
__global__ void transpose_v_kernel(const bf16_t* __restrict__ Vb, bf16_t* __restrict__ Vt) {
  size_t idx = (size_t)blockIdx.x * 256 + threadIdx.x;  // over B*T*D
  int dc = (int)(idx % kDH);
  size_t r = idx / kDH;
  int hh = (int)(r % kH); r /= kH;
  int t  = (int)(r % kT);
  int b  = (int)(r / kT);
  Vt[(((size_t)(b * kH + hh) * kDH) + dc) * kT + t] = Vb[idx];
}

// ---------------------------------------------------------------------------
// LayerNorm: fp32 [rows, kD] -> bf16 [rows, kD]
// ---------------------------------------------------------------------------
__global__ __launch_bounds__(256)
void layernorm_kernel(const float* __restrict__ x, const float* __restrict__ g,
                      const float* __restrict__ bb, bf16_t* __restrict__ out) {
  __shared__ float red[256];
  const int row = blockIdx.x, tid = threadIdx.x;
  const float* xr = x + (size_t)row * kD;
  float v[4];
  float s = 0.f;
#pragma unroll
  for (int q = 0; q < 4; ++q) { v[q] = xr[tid + q * 256]; s += v[q]; }
  red[tid] = s; __syncthreads();
  for (int off = 128; off > 0; off >>= 1) {
    if (tid < off) red[tid] += red[tid + off];
    __syncthreads();
  }
  const float mean = red[0] * (1.0f / kD);
  __syncthreads();
  float s2 = 0.f;
#pragma unroll
  for (int q = 0; q < 4; ++q) { float d = v[q] - mean; s2 += d * d; }
  red[tid] = s2; __syncthreads();
  for (int off = 128; off > 0; off >>= 1) {
    if (tid < off) red[tid] += red[tid + off];
    __syncthreads();
  }
  const float inv = rsqrtf(red[0] * (1.0f / kD) + 1e-5f);
  bf16_t* o = out + (size_t)row * kD;
#pragma unroll
  for (int q = 0; q < 4; ++q) {
    int c = tid + q * 256;
    o[c] = (bf16_t)((v[q] - mean) * inv * g[c] + bb[c]);
  }
}

// ---------------------------------------------------------------------------
// Main bf16 WMMA GEMM: C[M,N] = A[M,K] @ Bt[N,K]^T (+bias) (+epilogue)
// EPI: 0 = fp32 + bias; 1 = bf16 + bias; 2 = bf16 + bias + exact GELU;
//      3 = fp32 + bias + residual
// Block: 128 threads (4 waves, 2x2), tile 128x128, K-step 32.
// Tiles staged via async global->LDS (double-buffered, overlapped with WMMA).
// ---------------------------------------------------------------------------
template <int EPI>
__global__ __launch_bounds__(128)
void gemm_kernel(const bf16_t* __restrict__ A, const bf16_t* __restrict__ Bt,
                 const float* __restrict__ bias, const float* __restrict__ resid,
                 void* __restrict__ outp, int M, int N, int K) {
  __shared__ bf16_t As[2][128 * LDSSTRIDE];
  __shared__ bf16_t Bs[2][128 * LDSSTRIDE];
  const int tid  = threadIdx.x;
  const int lane = tid & 31;
  const int wave = tid >> 5;
  const int wr = (wave >> 1) * 64;
  const int wc = (wave & 1) * 64;
  const int rowBase = blockIdx.y * 128;
  const int colBase = blockIdx.x * 128;

  // thread t owns tile row t of A and of Bt (64B per K-slice)
  const bf16_t* aRow = A  + (size_t)(rowBase + tid) * K;
  const bf16_t* bRow = Bt + (size_t)(colBase + tid) * K;
  const unsigned ldsA0 = (unsigned)(size_t)(&As[0][tid * LDSSTRIDE]);
  const unsigned ldsA1 = (unsigned)(size_t)(&As[1][tid * LDSSTRIDE]);
  const unsigned ldsB0 = (unsigned)(size_t)(&Bs[0][tid * LDSSTRIDE]);
  const unsigned ldsB1 = (unsigned)(size_t)(&Bs[1][tid * LDSSTRIDE]);

  const v8f vzero = {0.f, 0.f, 0.f, 0.f, 0.f, 0.f, 0.f, 0.f};
  v8f acc[4][4];
#pragma unroll
  for (int i = 0; i < 4; ++i)
#pragma unroll
    for (int j = 0; j < 4; ++j) acc[i][j] = vzero;

  // prologue: stage slice 0 into buffer 0
  async_stage_64B(aRow, ldsA0);
  async_stage_64B(bRow, ldsB0);
  async_wait0();
  __syncthreads();

  const int nIter = K >> 5;
  for (int it = 0; it < nIter; ++it) {
    const int cur = it & 1;
    // prefetch next K-slice into the other buffer (overlaps with WMMA below)
    if (it + 1 < nIter) {
      async_stage_64B(aRow + (it + 1) * 32, cur ? ldsA0 : ldsA1);
      async_stage_64B(bRow + (it + 1) * 32, cur ? ldsB0 : ldsB1);
    }

    const bf16_t* as = As[cur];
    const bf16_t* bs = Bs[cur];
    v16bf aF[4], bF[4];
#pragma unroll
    for (int i = 0; i < 4; ++i) aF[i] = frag_a_ptr(as, LDSSTRIDE, wr + i * 16, 0, lane);
#pragma unroll
    for (int j = 0; j < 4; ++j) bF[j] = frag_b_ptr(bs, LDSSTRIDE, wc + j * 16, 0, lane);
#pragma unroll
    for (int i = 0; i < 4; ++i)
#pragma unroll
      for (int j = 0; j < 4; ++j) acc[i][j] = wmma_bf16(aF[i], bF[j], acc[i][j]);

    async_wait0();     // prefetched slice landed in LDS
    __syncthreads();   // visible to all waves; buffer `cur` free for re-stage
  }

  // epilogue: C row = v + laneHi*8, col = lane&15
#pragma unroll
  for (int i = 0; i < 4; ++i) {
#pragma unroll
    for (int j = 0; j < 4; ++j) {
      const int c  = colBase + wc + j * 16 + (lane & 15);
      const int r0 = rowBase + wr + i * 16 + ((lane >> 4) << 3);
      const float bv = bias ? bias[c] : 0.0f;
#pragma unroll
      for (int v = 0; v < 8; ++v) {
        const int r = r0 + v;
        float val = acc[i][j][v] + bv;
        const size_t idx = (size_t)r * N + c;
        if constexpr (EPI == 0) {
          ((float*)outp)[idx] = val;
        } else if constexpr (EPI == 1) {
          ((bf16_t*)outp)[idx] = (bf16_t)val;
        } else if constexpr (EPI == 2) {
          float gl = 0.5f * val * (1.0f + erff(val * 0.70710678118654752f));
          ((bf16_t*)outp)[idx] = (bf16_t)gl;
        } else {
          ((float*)outp)[idx] = val + resid[idx];
        }
      }
    }
  }
}

// ---------------------------------------------------------------------------
// Attention scores: S[z, i, j] = (Q_i . K_j)/8 with causal-window mask
// grid (4, 2, kZ): N=512 in 4 col tiles, M=256 in 2 row tiles.
// ---------------------------------------------------------------------------
__global__ __launch_bounds__(128)
void attn_scores_kernel(const bf16_t* __restrict__ Qb, const bf16_t* __restrict__ Kb,
                        float* __restrict__ S) {
  const int z = blockIdx.z;
  const int b    = z / (kH * kNB);
  const int hh   = (z / kNB) % kH;
  const int nblk = z % kNB;
  const int lane = threadIdx.x & 31;
  const int wave = threadIdx.x >> 5;
  const int wr = (wave >> 1) * 64;
  const int wc = (wave & 1) * 64;
  const int qBase = blockIdx.y * 128;
  const int jBase = blockIdx.x * 128;
  const int minJ = (nblk == 0) ? kW : 0;  // keys with pos<0 are zero/masked

  const bf16_t* Qp = Qb + ((size_t)(b * kT + nblk * kW) * kD + hh * kDH);
  // virtual window start (j=0 -> key pos nblk*kW - kW); never dereferenced for j<minJ
  const bf16_t* Kp = Kb + ((long long)(b * kT + nblk * kW - kW) * kD + hh * kDH);

  const v8f vzero = {0.f, 0.f, 0.f, 0.f, 0.f, 0.f, 0.f, 0.f};
  v8f acc[4][4];
#pragma unroll
  for (int i = 0; i < 4; ++i)
#pragma unroll
    for (int j = 0; j < 4; ++j) acc[i][j] = vzero;

#pragma unroll
  for (int k0 = 0; k0 < kDH; k0 += 32) {
    v16bf aF[4], bF[4];
#pragma unroll
    for (int i = 0; i < 4; ++i)
      aF[i] = frag_a_ptr(Qp, kD, qBase + wr + i * 16, k0, lane);
#pragma unroll
    for (int j = 0; j < 4; ++j)
      bF[j] = frag_b_rowpred(Kp, kD, jBase + wc + j * 16, k0, lane, minJ);
#pragma unroll
    for (int i = 0; i < 4; ++i)
#pragma unroll
      for (int j = 0; j < 4; ++j) acc[i][j] = wmma_bf16(aF[i], bF[j], acc[i][j]);
  }

  float* Sp = S + (size_t)z * kW * kW2;
#pragma unroll
  for (int i = 0; i < 4; ++i) {
#pragma unroll
    for (int j = 0; j < 4; ++j) {
      const int jc = jBase + wc + j * 16 + (lane & 15);
      const int r0 = qBase + wr + i * 16 + ((lane >> 4) << 3);
#pragma unroll
      for (int v = 0; v < 8; ++v) {
        const int iq = r0 + v;
        const bool valid = (jc > iq) && (jc <= iq + kW) && (jc >= minJ);
        Sp[(size_t)iq * kW2 + jc] = valid ? acc[i][j][v] * 0.125f : -1.0e30f;
      }
    }
  }
}

// ---------------------------------------------------------------------------
// Row softmax over 512 keys -> bf16 probabilities
// ---------------------------------------------------------------------------
__global__ __launch_bounds__(128)
void softmax_kernel(const float* __restrict__ S, bf16_t* __restrict__ P) {
  __shared__ float red[128];
  const size_t row = blockIdx.x;
  const int tid = threadIdx.x;
  const float* s = S + row * kW2;
  float v[4];
  float m = -3.0e38f;
#pragma unroll
  for (int q = 0; q < 4; ++q) { v[q] = s[tid + q * 128]; m = fmaxf(m, v[q]); }
  red[tid] = m; __syncthreads();
  for (int off = 64; off > 0; off >>= 1) {
    if (tid < off) red[tid] = fmaxf(red[tid], red[tid + off]);
    __syncthreads();
  }
  m = red[0];
  __syncthreads();
  float e[4];
  float sum = 0.f;
#pragma unroll
  for (int q = 0; q < 4; ++q) { e[q] = __expf(v[q] - m); sum += e[q]; }
  red[tid] = sum; __syncthreads();
  for (int off = 64; off > 0; off >>= 1) {
    if (tid < off) red[tid] += red[tid + off];
    __syncthreads();
  }
  const float inv = 1.0f / red[0];
  bf16_t* p = P + row * kW2;
#pragma unroll
  for (int q = 0; q < 4; ++q) p[tid + q * 128] = (bf16_t)(e[q] * inv);
}

// ---------------------------------------------------------------------------
// PV: ctx[b,t,h,dh] = P[z] @ Vwin ; one block per z, M=256 (4 waves x 64), N=64
// ---------------------------------------------------------------------------
__global__ __launch_bounds__(128)
void attn_pv_kernel(const bf16_t* __restrict__ P, const bf16_t* __restrict__ Vt,
                    bf16_t* __restrict__ ctx) {
  const int z = blockIdx.x;
  const int b    = z / (kH * kNB);
  const int hh   = (z / kNB) % kH;
  const int nblk = z % kNB;
  const int lane = threadIdx.x & 31;
  const int wave = threadIdx.x >> 5;
  const int wr = wave * 64;
  const long long winStart = (long long)nblk * kW - kW;
  const bf16_t* Pz = P + (size_t)z * kW * kW2;
  const bf16_t* Vz = Vt + ((size_t)(b * kH + hh) * kDH) * kT;

  const v8f vzero = {0.f, 0.f, 0.f, 0.f, 0.f, 0.f, 0.f, 0.f};
  v8f acc[4][4];
#pragma unroll
  for (int i = 0; i < 4; ++i)
#pragma unroll
    for (int j = 0; j < 4; ++j) acc[i][j] = vzero;

  for (int k0 = 0; k0 < kW2; k0 += 32) {
    v16bf aF[4], bF[4];
#pragma unroll
    for (int i = 0; i < 4; ++i)
      aF[i] = frag_a_ptr(Pz, kW2, wr + i * 16, k0, lane);
#pragma unroll
    for (int j = 0; j < 4; ++j) {
      const int dc = j * 16 + (lane & 15);
      const bf16_t* p = Vz + (size_t)dc * kT;
      const int kk = k0 + ((lane >> 4) << 4);   // 16 contiguous keys per lane
      const long long p0 = winStart + kk;
      const long long p1 = winStart + kk + 8;
      FragU f;
      f.q[0] = (p0 >= 0) ? *reinterpret_cast<const uint4*>(p + p0) : make_uint4(0u, 0u, 0u, 0u);
      f.q[1] = (p1 >= 0) ? *reinterpret_cast<const uint4*>(p + p1) : make_uint4(0u, 0u, 0u, 0u);
      bF[j] = f.v;
    }
#pragma unroll
    for (int i = 0; i < 4; ++i)
#pragma unroll
      for (int j = 0; j < 4; ++j) acc[i][j] = wmma_bf16(aF[i], bF[j], acc[i][j]);
  }

#pragma unroll
  for (int i = 0; i < 4; ++i) {
#pragma unroll
    for (int j = 0; j < 4; ++j) {
      const int c  = j * 16 + (lane & 15);
      const int r0 = wr + i * 16 + ((lane >> 4) << 3);
#pragma unroll
      for (int v = 0; v < 8; ++v) {
        const int t = nblk * kW + r0 + v;
        ctx[((size_t)(b * kT + t) * kD) + hh * kDH + c] = (bf16_t)acc[i][j][v];
      }
    }
  }
}

// ---------------------------------------------------------------------------
// SSM scan: s_t = tanh(A)*s_{t-1} + u_t  (512 independent channels)
// ---------------------------------------------------------------------------
__global__ __launch_bounds__(256)
void ssm_scan_kernel(const float* __restrict__ u, const float* __restrict__ Aw,
                     const float* __restrict__ s0, bf16_t* __restrict__ st,
                     float* __restrict__ newstate) {
  const int b = blockIdx.x;
  const int n = threadIdx.x;
  const float alpha = tanhf(Aw[n]);
  float s = s0[b * kN + n];
  const float* up = u + (size_t)b * kT * kN + n;
  bf16_t* sp = st + (size_t)b * kT * kN + n;
  for (int t = 0; t < kT; ++t) {
    s = alpha * s + up[(size_t)t * kN];
    sp[(size_t)t * kN] = (bf16_t)s;
  }
  newstate[b * kN + n] = s;
}

// ---------------------------------------------------------------------------
// Gated fusion + residual: x1 = x + sig(g)*y_attn + (1-sig(g))*y_ssm
// ---------------------------------------------------------------------------
__global__ __launch_bounds__(256)
void fuse_kernel(const float* __restrict__ x, const float* __restrict__ graw,
                 const float* __restrict__ yattn, const float* __restrict__ yssm,
                 float* __restrict__ x1) {
  const size_t idx = (size_t)blockIdx.x * 256 + threadIdx.x;
  const float gg = 1.0f / (1.0f + __expf(-graw[idx]));
  x1[idx] = x[idx] + gg * yattn[idx] + (1.0f - gg) * yssm[idx];
}

// ---------------------------------------------------------------------------
extern "C" void kernel_launch(void* const* d_in, const int* in_sizes, int n_in,
                              void* d_out, int out_size, void* d_ws, size_t ws_size,
                              hipStream_t stream) {
  const float* x    = (const float*)d_in[0];
  const float* ssm0 = (const float*)d_in[1];
  const float* ln1g = (const float*)d_in[2];
  const float* ln1b = (const float*)d_in[3];
  const float* ln2g = (const float*)d_in[4];
  const float* ln2b = (const float*)d_in[5];
  const float* wq   = (const float*)d_in[6];
  const float* bq   = (const float*)d_in[7];
  const float* wk   = (const float*)d_in[8];
  const float* bk   = (const float*)d_in[9];
  const float* wv   = (const float*)d_in[10];
  const float* bv   = (const float*)d_in[11];
  const float* wo   = (const float*)d_in[12];
  const float* bo   = (const float*)d_in[13];
  const float* wg   = (const float*)d_in[14];
  const float* bg   = (const float*)d_in[15];
  const float* Aw   = (const float*)d_in[16];
  const float* Bw   = (const float*)d_in[17];
  const float* Cw   = (const float*)d_in[18];
  const float* w1   = (const float*)d_in[19];
  const float* b1   = (const float*)d_in[20];
  const float* w2   = (const float*)d_in[21];
  const float* b2   = (const float*)d_in[22];
  (void)in_sizes; (void)n_in; (void)out_size; (void)ws_size;

  // Workspace carve (bytes, 256B aligned per buffer)
  char* wp = (char*)d_ws;
  auto carve = [&](size_t bytes) -> char* {
    char* p = wp;
    wp += (bytes + 255) & ~(size_t)255;
    return p;
  };
  bf16_t* wqT = (bf16_t*)carve((size_t)kD * kD * 2);
  bf16_t* wkT = (bf16_t*)carve((size_t)kD * kD * 2);
  bf16_t* wvT = (bf16_t*)carve((size_t)kD * kD * 2);
  bf16_t* wgT = (bf16_t*)carve((size_t)kD * kD * 2);
  bf16_t* woT = (bf16_t*)carve((size_t)kD * kD * 2);
  bf16_t* w1T = (bf16_t*)carve((size_t)kD * kDFF * 2);
  bf16_t* w2T = (bf16_t*)carve((size_t)kDFF * kD * 2);
  bf16_t* BwT = (bf16_t*)carve((size_t)kD * kN * 2);
  bf16_t* CwT = (bf16_t*)carve((size_t)kN * kD * 2);
  bf16_t* xn  = (bf16_t*)carve((size_t)kBT * kD * 2);
  bf16_t* Qb  = (bf16_t*)carve((size_t)kBT * kD * 2);
  bf16_t* Kb  = (bf16_t*)carve((size_t)kBT * kD * 2);
  bf16_t* Vb  = (bf16_t*)carve((size_t)kBT * kD * 2);
  bf16_t* Vt  = (bf16_t*)carve((size_t)kBT * kD * 2);
  float*  graw  = (float*)carve((size_t)kBT * kD * 4);
  float*  Sbuf  = (float*)carve((size_t)kZ * kW * kW2 * 4);   // 128 MB
  bf16_t* Pbuf  = (bf16_t*)carve((size_t)kZ * kW * kW2 * 2);  // 64 MB
  bf16_t* ctx   = (bf16_t*)carve((size_t)kBT * kD * 2);
  float*  yattn = (float*)carve((size_t)kBT * kD * 4);
  float*  ubuf  = (float*)carve((size_t)kBT * kN * 4);
  bf16_t* stb   = (bf16_t*)carve((size_t)kBT * kN * 2);
  float*  yssm  = (float*)carve((size_t)kBT * kD * 4);
  float*  x1    = (float*)carve((size_t)kBT * kD * 4);
  bf16_t* x2n   = (bf16_t*)carve((size_t)kBT * kD * 2);
  bf16_t* h1    = (bf16_t*)carve((size_t)kBT * kDFF * 2);

  float* xout     = (float*)d_out;
  float* newstate = xout + (size_t)kBT * kD;

  // 1) weights -> bf16 [N,K]
  {
    int blkDD  = (kD * kD) / 256;
    int blkDF  = (kD * kDFF) / 256;
    int blkDN  = (kD * kN) / 256;
    convert_wt_kernel<<<blkDD, 256, 0, stream>>>(wq, wqT, kD, kD);
    convert_wt_kernel<<<blkDD, 256, 0, stream>>>(wk, wkT, kD, kD);
    convert_wt_kernel<<<blkDD, 256, 0, stream>>>(wv, wvT, kD, kD);
    convert_wt_kernel<<<blkDD, 256, 0, stream>>>(wg, wgT, kD, kD);
    convert_wt_kernel<<<blkDD, 256, 0, stream>>>(wo, woT, kD, kD);
    convert_wt_kernel<<<blkDF, 256, 0, stream>>>(w1, w1T, kD, kDFF);
    convert_wt_kernel<<<blkDF, 256, 0, stream>>>(w2, w2T, kDFF, kD);
    convert_wt_kernel<<<blkDN, 256, 0, stream>>>(Bw, BwT, kD, kN);
    convert_wt_kernel<<<blkDN, 256, 0, stream>>>(Cw, CwT, kN, kD);
  }

  // 2) LN1
  layernorm_kernel<<<kBT, 256, 0, stream>>>(x, ln1g, ln1b, xn);

  // 3) projections (WMMA GEMMs, async-LDS staged)
  {
    dim3 g(kD / 128, kBT / 128);
    gemm_kernel<1><<<g, 128, 0, stream>>>(xn, wqT, bq, nullptr, Qb,   kBT, kD, kD);
    gemm_kernel<1><<<g, 128, 0, stream>>>(xn, wkT, bk, nullptr, Kb,   kBT, kD, kD);
    gemm_kernel<1><<<g, 128, 0, stream>>>(xn, wvT, bv, nullptr, Vb,   kBT, kD, kD);
    gemm_kernel<0><<<g, 128, 0, stream>>>(xn, wgT, bg, nullptr, graw, kBT, kD, kD);
  }

  // 4) V transpose for PV
  transpose_v_kernel<<<(kBT * kD) / 256, 256, 0, stream>>>(Vb, Vt);

  // 5) attention scores, softmax, PV
  attn_scores_kernel<<<dim3(kW2 / 128, kW / 128, kZ), 128, 0, stream>>>(Qb, Kb, Sbuf);
  softmax_kernel<<<kZ * kW, 128, 0, stream>>>(Sbuf, Pbuf);
  attn_pv_kernel<<<kZ, 128, 0, stream>>>(Pbuf, Vt, ctx);

  // 6) output projection
  gemm_kernel<0><<<dim3(kD / 128, kBT / 128), 128, 0, stream>>>(ctx, woT, bo, nullptr, yattn, kBT, kD, kD);

  // 7) SSM path
  gemm_kernel<0><<<dim3(kN / 128, kBT / 128), 128, 0, stream>>>(xn, BwT, nullptr, nullptr, ubuf, kBT, kN, kD);
  ssm_scan_kernel<<<kB, kN, 0, stream>>>(ubuf, Aw, ssm0, stb, newstate);
  gemm_kernel<0><<<dim3(kD / 128, kBT / 128), 128, 0, stream>>>(stb, CwT, nullptr, nullptr, yssm, kBT, kD, kN);

  // 8) gated fusion + residual
  fuse_kernel<<<(kBT * kD) / 256, 256, 0, stream>>>(x, graw, yattn, yssm, x1);

  // 9) LN2 + MLP (GELU fused into GEMM1 epilogue, residual fused into GEMM2)
  layernorm_kernel<<<kBT, 256, 0, stream>>>(x1, ln2g, ln2b, x2n);
  gemm_kernel<2><<<dim3(kDFF / 128, kBT / 128), 128, 0, stream>>>(x2n, w1T, b1, nullptr, h1, kBT, kDFF, kD);
  gemm_kernel<3><<<dim3(kD / 128, kBT / 128), 128, 0, stream>>>(h1, w2T, b2, x1, xout, kBT, kD, kDFF);
}
